// FP8Linear_63995012710474
// MI455X (gfx1250) — 8-bit matrix operand order repaired and checked by reading, compiled, not run on hardware
//
#include <hip/hip_runtime.h>
#include <stdint.h>

typedef int   v4i  __attribute__((ext_vector_type(4)));
typedef int   v16i __attribute__((ext_vector_type(16)));
typedef float v4f  __attribute__((ext_vector_type(4)));
typedef float v8f  __attribute__((ext_vector_type(8)));

#define TILE_M 128
#define TILE_N 128
#define KSTEP  128

__global__ void quantize_fp8(const float* __restrict__ src,
                             uint8_t* __restrict__ dst,
                             long long n4) {
    long long i      = (long long)blockIdx.x * blockDim.x + threadIdx.x;
    long long stride = (long long)gridDim.x * blockDim.x;
    const v4f* s4 = (const v4f*)src;
    int* d4 = (int*)dst;
    for (; i < n4; i += stride) {
        v4f f = __builtin_nontemporal_load(s4 + i);
        int lo     = __builtin_amdgcn_cvt_pk_fp8_f32(f[0], f[1], 0,  false);
        int packed = __builtin_amdgcn_cvt_pk_fp8_f32(f[2], f[3], lo, true);
        d4[i] = packed;
    }
}

__global__ __launch_bounds__(256)
void fp8_gemm_wmma(const uint8_t* __restrict__ xq,
                   const uint8_t* __restrict__ wq,
                   const float*   __restrict__ bias,
                   const float*   __restrict__ scale_a,
                   const float*   __restrict__ scale_b,
                   float*         __restrict__ out,
                   int M, int N, int K) {
    const int lane = threadIdx.x & 31;
    const int wave = threadIdx.x >> 5;
    const int wm   = wave & 3;
    const int wn   = wave >> 2;
    const int r    = lane & 15;
    const int h    = lane >> 4;

    const int m0 = blockIdx.y * TILE_M + wm * 32;
    const int n0 = blockIdx.x * TILE_N + wn * 64;

    v8f acc[2][4];
#pragma unroll
    for (int i = 0; i < 2; ++i)
#pragma unroll
        for (int j = 0; j < 4; ++j)
#pragma unroll
            for (int v = 0; v < 8; ++v)
                acc[i][j][v] = 0.0f;

    const uint8_t* arow = xq + (size_t)(m0 + r) * (size_t)K;
    const uint8_t* brow = wq + (size_t)(n0 + r) * (size_t)K;

    for (int k0 = 0; k0 < K; k0 += KSTEP) {
        __builtin_prefetch(arow + k0 + KSTEP, 0, 1);
        __builtin_prefetch(brow + k0 + KSTEP, 0, 1);

        v16i A[2];
#pragma unroll
        for (int i = 0; i < 2; ++i) {
            const uint8_t* pa = arow + (size_t)i * 16u * (size_t)K + k0 + h * 16;
#pragma unroll
            for (int halfk = 0; halfk < 2; ++halfk) {
#pragma unroll
                for (int g = 0; g < 4; ++g) {
                    uint64_t d = *(const uint64_t*)(pa + halfk * 64 + (g >> 1) * 32 + (g & 1) * 8);
                    A[i][halfk * 8 + g * 2 + 0] = (int)(uint32_t)(d);
                    A[i][halfk * 8 + g * 2 + 1] = (int)(uint32_t)(d >> 32);
                }
            }
        }

#pragma unroll
        for (int j = 0; j < 4; ++j) {
            const uint8_t* pb = brow + (size_t)j * 16u * (size_t)K + k0 + h * 16;
            v16i B;
#pragma unroll
            for (int q = 0; q < 4; ++q) {
                v4i d = *(const v4i*)(pb + q * 32);
                B[q * 4 + 0] = d[0];
                B[q * 4 + 1] = d[1];
                B[q * 4 + 2] = d[2];
                B[q * 4 + 3] = d[3];
            }
#pragma unroll
            for (int i = 0; i < 2; ++i) {
                acc[i][j] = __builtin_amdgcn_wmma_f32_16x16x128_fp8_fp8(
                    A[i], B, (short)0, acc[i][j],
                     false,  false);
            }
        }
    }

    const float s = scale_a[0] * scale_b[0];
#pragma unroll
    for (int j = 0; j < 4; ++j) {
        const int col = n0 + j * 16 + r;
        const float bv = bias[col];
#pragma unroll
        for (int i = 0; i < 2; ++i) {
            float* po = out + (size_t)(m0 + i * 16 + h * 8) * (size_t)N + col;
#pragma unroll
            for (int v = 0; v < 8; ++v)
                __builtin_nontemporal_store(acc[i][j][v] * s + bv,
                                            po + (size_t)v * (size_t)N);
        }
    }
}

extern "C" void kernel_launch(void* const* d_in, const int* in_sizes, int n_in,
                              void* d_out, int out_size, void* d_ws, size_t ws_size,
                              hipStream_t stream) {
    const float* x   = (const float*)d_in[0];
    const float* w   = (const float*)d_in[1];
    const float* b   = (const float*)d_in[2];
    const float* sa  = (const float*)d_in[3];
    const float* sb  = (const float*)d_in[4];
    float* out = (float*)d_out;

    const long long xk = (long long)in_sizes[0];
    const long long wk = (long long)in_sizes[1];
    const int N = in_sizes[2];
    const int K = (int)(wk / N);
    const int M = (int)(xk / K);

    uint8_t* xq = (uint8_t*)d_ws;
    uint8_t* wq = xq + xk;

    {
        long long n4 = xk / 4;
        int blocks = (int)((n4 + 255) / 256);
        quantize_fp8<<<blocks, 256, 0, stream>>>(x, xq, n4);
    }
    {
        long long n4 = wk / 4;
        int blocks = (int)((n4 + 255) / 256);
        quantize_fp8<<<blocks, 256, 0, stream>>>(w, wq, n4);
    }

    dim3 grid((unsigned)(N / TILE_N), (unsigned)(M / TILE_M));
    fp8_gemm_wmma<<<grid, 256, 0, stream>>>(xq, wq, b, sa, sb, out, M, N, K);
}
